// MultiHeadSelfAttention_2954937500392
// MI455X (gfx1250) — compile-verified, matrix operand order checked by reading, not run on hardware
//
#include <hip/hip_runtime.h>
#include <hip/hip_bf16.h>

// ---------------------------------------------------------------------------
// MI455X (gfx1250) multi-head self-attention, f16 WMMA path, f32 accumulate.
// B=2, S=2048, D=1024, H=16, P=64.
// ~69 GFLOP total vs ~50 MB minimum HBM traffic -> compute-bound: everything
// runs on v_wmma_f32_16x16x32_f16 with f16 intermediates held in workspace
// (Q/K/V set = 24 MB, fully L2-resident at 192 MB).
// ---------------------------------------------------------------------------

#define SEQ   2048
#define DIM   1024
#define NH    16
#define HD    64
#define BATCH 2
#define ROWS  (BATCH * SEQ)          // 4096

typedef _Float16 h8  __attribute__((ext_vector_type(8)));
typedef _Float16 h16 __attribute__((ext_vector_type(16)));
typedef float    f8  __attribute__((ext_vector_type(8)));

// Assemble a 16x32 f16 A/B WMMA fragment (16 halves per lane) from two
// contiguous 16-byte chunks per the CDNA5 16-bit operand layout:
//   lane L: row/col = L&15, kbase = 8*(L>>4);
//   elems 0..7  -> K = kbase+e      (first 16B chunk)
//   elems 8..15 -> K = 16+kbase+e-8 (second 16B chunk, +16 halves away)
__device__ __forceinline__ h16 ld_frag(const _Float16* lo, const _Float16* hi) {
    h8 a = *(const h8*)lo;
    h8 b = *(const h8*)hi;
    return __builtin_shufflevector(a, b, 0,1,2,3,4,5,6,7,8,9,10,11,12,13,14,15);
}

__device__ __forceinline__ f8 f8zero() {
    f8 z;
#pragma unroll
    for (int i = 0; i < 8; ++i) z[i] = 0.0f;
    return z;
}

__device__ __forceinline__ f8 wmma_f16(h16 a, h16 b, f8 c) {
    return __builtin_amdgcn_wmma_f32_16x16x32_f16(false, a, false, b,
                                                  (short)0, c, false, false);
}

// Row M of a 16x16 C tile lives in a 16-lane half-wave; xor masks < 16 never
// cross the half boundary, so plain shfl_xor reductions stay within a row.
__device__ __forceinline__ float rowmax16(float v) {
#pragma unroll
    for (int off = 1; off < 16; off <<= 1) v = fmaxf(v, __shfl_xor(v, off));
    return v;
}
__device__ __forceinline__ float rowsum16(float v) {
#pragma unroll
    for (int off = 1; off < 16; off <<= 1) v += __shfl_xor(v, off);
    return v;
}

// ---------------------------------------------------------------------------
// Kernel 1: per-weight projection GEMM.
//   X[4096x1024] (f32) x W[1024x1024] (f32) + bias -> f16 workspace.
//   vtranspose==0: out[((b*NH+h)*SEQ + s)*HD + p]   (Q, K head-major)
//   vtranspose==1: out[((b*NH+h)*HD  + p)*SEQ + s]  (V transposed for PV)
// Block = 256 threads (8 waves, 2(M) x 4(N)); block tile 32 x 128;
// wave tile 16 x 32 -> 2 WMMA per k-step sharing one A fragment.
// LDS = 10 KB (32x32 X chunk + 128x32 transposed W chunk).
// ---------------------------------------------------------------------------
__global__ __launch_bounds__(256)
void qkv_proj_kernel(const float* __restrict__ X, const float* __restrict__ W,
                     const float* __restrict__ bias, _Float16* __restrict__ out,
                     int vtranspose) {
    __shared__ _Float16 Xs[32][32];     // 2 KB: X k-chunk, f16
    __shared__ _Float16 Wt[128][32];    // 8 KB: W k-chunk, transposed

    const int tid  = threadIdx.x;
    const int lane = tid & 31;
    const int wave = tid >> 5;
    const int mw   = wave >> 2;          // 0..1
    const int nw   = wave & 3;           // 0..3
    const int m0   = blockIdx.x * 32;
    const int n0   = blockIdx.y * 128;

    const int mrow = lane & 15;
    const int hi   = lane >> 4;
    const int kb   = hi * 8;

    f8 acc0 = f8zero(), acc1 = f8zero();

    for (int kk = 0; kk < DIM; kk += 32) {
        // Cooperative, coalesced f32->f16 staging of X[m0:m0+32, kk:kk+32].
#pragma unroll
        for (int i = 0; i < 4; ++i) {
            int e = tid + 256 * i;            // 0 .. 1023
            int r = e >> 5, c = e & 31;
            Xs[r][c] = (_Float16)X[(m0 + r) * DIM + kk + c];
        }
        // Cooperative, coalesced staging of W[kk:kk+32, n0:n0+128] (transposed).
#pragma unroll 4
        for (int i = 0; i < 16; ++i) {
            int e = tid + 256 * i;            // 0 .. 4095
            int r = e >> 7, c = e & 127;
            Wt[c][r] = (_Float16)W[(kk + r) * DIM + n0 + c];
        }
        __syncthreads();

        h16 a  = ld_frag(&Xs[mw * 16 + mrow][kb],      &Xs[mw * 16 + mrow][16 + kb]);
        h16 b0 = ld_frag(&Wt[nw * 32 + mrow][kb],      &Wt[nw * 32 + mrow][16 + kb]);
        h16 b1 = ld_frag(&Wt[nw * 32 + 16 + mrow][kb], &Wt[nw * 32 + 16 + mrow][16 + kb]);
        acc0 = wmma_f16(a, b0, acc0);         // same A operand back-to-back
        acc1 = wmma_f16(a, b1, acc1);
        __syncthreads();
    }

    // Epilogue: bias + f16 convert + scatter into head-major / transposed-V.
#pragma unroll
    for (int t = 0; t < 2; ++t) {
        f8  acc   = t ? acc1 : acc0;
        int nglob = n0 + nw * 32 + t * 16 + mrow;
        int hh    = nglob >> 6;
        int pp    = nglob & 63;
        float bval = bias[nglob];
#pragma unroll
        for (int r = 0; r < 8; ++r) {
            int srow = m0 + mw * 16 + r + 8 * hi;   // 0 .. 4095
            int bb   = srow >> 11;                  // / SEQ
            int ss   = srow & (SEQ - 1);
            float v  = acc[r] + bval;
            if (vtranspose)
                out[(((bb * NH + hh) * HD) + pp) * SEQ + ss] = (_Float16)v;
            else
                out[(((bb * NH + hh) * SEQ) + ss) * HD + pp] = (_Float16)v;
        }
    }
}

// ---------------------------------------------------------------------------
// Kernel 2: flash attention. One wave (single-wave workgroup -> barriers are
// NOPs per ISA 3.1) per (b, h, 16-query tile). Streaming online softmax over
// 32-key chunks: 4 WMMA (QK^T, K-dim=64) + 4 WMMA (PV, N=64) per chunk.
// 16-bit WMMA is TRANS-class, so the softmax VALU work co-executes with it.
// ---------------------------------------------------------------------------
__global__ __launch_bounds__(32)
void attn_kernel(const _Float16* __restrict__ Q, const _Float16* __restrict__ K,
                 const _Float16* __restrict__ Vt, _Float16* __restrict__ O) {
    __shared__ _Float16 Plds[16][32];         // probs: C-layout -> A-layout bounce

    const int lane = threadIdx.x;
    const int q0   = blockIdx.x * 16;
    const int h    = blockIdx.y;
    const int b    = blockIdx.z;
    const int mrow = lane & 15;
    const int hi   = lane >> 4;
    const int kb   = hi * 8;

    const _Float16* qh = Q  + (size_t)(b * NH + h) * SEQ * HD;
    const _Float16* kh = K  + (size_t)(b * NH + h) * SEQ * HD;
    const _Float16* vh = Vt + (size_t)(b * NH + h) * HD * SEQ;

    // Q fragments for this tile (resident for the whole loop).
    const _Float16* qr = qh + (q0 + mrow) * HD;
    h16 aQ0 = ld_frag(qr + kb,      qr + 16 + kb);   // p = 0..31
    h16 aQ1 = ld_frag(qr + 32 + kb, qr + 48 + kb);   // p = 32..63

    f8 o[4];
#pragma unroll
    for (int t = 0; t < 4; ++t) o[t] = f8zero();

    float rmax[8], rsum[8];
#pragma unroll
    for (int r = 0; r < 8; ++r) { rmax[r] = -1e30f; rsum[r] = 0.0f; }

    const float SC = 0.125f * 1.44269504088896f;     // 1/sqrt(64) * log2(e)

    for (int j0 = 0; j0 < SEQ; j0 += 32) {
        if (j0 + 32 < SEQ) {                          // global_prefetch_b8
            __builtin_prefetch(kh + (j0 + 32 + mrow) * HD, 0, 1);
            __builtin_prefetch(vh + mrow * SEQ + j0 + 32, 0, 1);
        }

        // ---- scores: S[16 x 32] = Q (16x64) . K^T ----
        const _Float16* k0 = kh + (j0 + mrow) * HD;       // keys j0+0 .. +15
        const _Float16* k1 = kh + (j0 + 16 + mrow) * HD;  // keys j0+16 .. +31
        h16 b00 = ld_frag(k0 + kb,      k0 + 16 + kb);
        h16 b01 = ld_frag(k0 + 32 + kb, k0 + 48 + kb);
        h16 b10 = ld_frag(k1 + kb,      k1 + 16 + kb);
        h16 b11 = ld_frag(k1 + 32 + kb, k1 + 48 + kb);

        f8 c0 = f8zero(), c1 = f8zero();
        c0 = wmma_f16(aQ0, b00, c0);  c0 = wmma_f16(aQ1, b01, c0);
        c1 = wmma_f16(aQ0, b10, c1);  c1 = wmma_f16(aQ1, b11, c1);

        // ---- online softmax update (per C-tile row r) ----
        float p0[8], p1[8];
#pragma unroll
        for (int r = 0; r < 8; ++r) {
            float t0 = c0[r] * SC, t1 = c1[r] * SC;
            float mx = rowmax16(fmaxf(t0, t1));
            float mnew = fmaxf(rmax[r], mx);
            float alpha = exp2f(rmax[r] - mnew);
            rmax[r] = mnew;
            float e0 = exp2f(t0 - mnew);
            float e1 = exp2f(t1 - mnew);
            rsum[r] = rsum[r] * alpha + rowsum16(e0 + e1);
            p0[r] = e0; p1[r] = e1;
#pragma unroll
            for (int t = 0; t < 4; ++t) o[t][r] *= alpha;
        }

        // ---- probs: C-layout -> LDS -> A-layout fragment ----
#pragma unroll
        for (int r = 0; r < 8; ++r) {
            Plds[r + 8 * hi][mrow]      = (_Float16)p0[r];
            Plds[r + 8 * hi][16 + mrow] = (_Float16)p1[r];
        }
        __syncthreads();                                // NOP (single wave)
        h16 aP = ld_frag(&Plds[mrow][kb], &Plds[mrow][16 + kb]);

        // ---- PV: out[16 x 64] += P (16x32) . V (32x64) ----
#pragma unroll
        for (int t = 0; t < 4; ++t) {
            const _Float16* vp = vh + (t * 16 + mrow) * SEQ + j0;
            h16 bv = ld_frag(vp + kb, vp + 16 + kb);    // contiguous: V transposed
            o[t] = wmma_f16(aP, bv, o[t]);
        }
        __syncthreads();                                // NOP (single wave)
    }

    // ---- normalize + store attn in concat layout [b, s, h*64+p] (f16) ----
#pragma unroll
    for (int r = 0; r < 8; ++r) {
        float inv = 1.0f / rsum[r];
        int srow  = q0 + r + 8 * hi;
        _Float16* orow = O + ((size_t)b * SEQ + srow) * DIM + h * HD;
#pragma unroll
        for (int t = 0; t < 4; ++t)
            orow[t * 16 + mrow] = (_Float16)(o[t][r] * inv);
    }
}

// ---------------------------------------------------------------------------
// Kernel 3: output projection  attn[4096x1024](f16) x Wo + bo -> f32 d_out.
// Same 32x128 block tile / 16x32 wave tile as kernel 1; A fragments come
// straight from global (already f16, contiguous 16B). Final stores are
// non-temporal (write-once stream, never re-read).
// ---------------------------------------------------------------------------
__global__ __launch_bounds__(256)
void out_proj_kernel(const _Float16* __restrict__ A, const float* __restrict__ W,
                     const float* __restrict__ bias, float* __restrict__ out) {
    __shared__ _Float16 Wt[128][32];

    const int tid  = threadIdx.x;
    const int lane = tid & 31;
    const int wave = tid >> 5;
    const int mw   = wave >> 2;
    const int nw   = wave & 3;
    const int m0   = blockIdx.x * 32;
    const int n0   = blockIdx.y * 128;
    const int mrow = lane & 15;
    const int hi   = lane >> 4;
    const int kb   = hi * 8;

    const _Float16* arow = A + (size_t)(m0 + mw * 16 + mrow) * DIM;
    f8 acc0 = f8zero(), acc1 = f8zero();

    for (int kk = 0; kk < DIM; kk += 32) {
#pragma unroll 4
        for (int i = 0; i < 16; ++i) {
            int e = tid + 256 * i;
            int r = e >> 7, c = e & 127;
            Wt[c][r] = (_Float16)W[(kk + r) * DIM + n0 + c];
        }
        __syncthreads();

        h16 a  = ld_frag(arow + kk + kb, arow + kk + 16 + kb);
        h16 b0 = ld_frag(&Wt[nw * 32 + mrow][kb],      &Wt[nw * 32 + mrow][16 + kb]);
        h16 b1 = ld_frag(&Wt[nw * 32 + 16 + mrow][kb], &Wt[nw * 32 + 16 + mrow][16 + kb]);
        acc0 = wmma_f16(a, b0, acc0);
        acc1 = wmma_f16(a, b1, acc1);
        __syncthreads();
    }

#pragma unroll
    for (int t = 0; t < 2; ++t) {
        f8  acc   = t ? acc1 : acc0;
        int nglob = n0 + nw * 32 + t * 16 + mrow;
        float bval = bias[nglob];
#pragma unroll
        for (int r = 0; r < 8; ++r) {
            size_t idx = (size_t)(m0 + mw * 16 + r + 8 * hi) * DIM + nglob;
            __builtin_nontemporal_store(acc[r] + bval, out + idx);
        }
    }
}

// ---------------------------------------------------------------------------
// Launch
// ---------------------------------------------------------------------------
extern "C" void kernel_launch(void* const* d_in, const int* in_sizes, int n_in,
                              void* d_out, int out_size, void* d_ws, size_t ws_size,
                              hipStream_t stream) {
    const float* x  = (const float*)d_in[0];
    const float* Wq = (const float*)d_in[1];
    const float* bq = (const float*)d_in[2];
    const float* Wk = (const float*)d_in[3];
    const float* bk = (const float*)d_in[4];
    const float* Wv = (const float*)d_in[5];
    const float* bv = (const float*)d_in[6];
    const float* Wo = (const float*)d_in[7];
    const float* bo = (const float*)d_in[8];

    _Float16* ws = (_Float16*)d_ws;
    const size_t N = (size_t)ROWS * DIM;     // 4,194,304 elements each
    _Float16* qw = ws;                       //  8 MB  Q  [b,h,s,p]
    _Float16* kw = ws + N;                   //  8 MB  K  [b,h,s,p]
    _Float16* vw = ws + 2 * N;               //  8 MB  V^T[b,h,p,s]
    _Float16* aw = ws + 3 * N;               //  8 MB  attn [b,s,d]

    dim3 gemm_grid(ROWS / 32, DIM / 128);    // 128 x 8
    qkv_proj_kernel<<<gemm_grid, 256, 0, stream>>>(x, Wq, bq, qw, 0);
    qkv_proj_kernel<<<gemm_grid, 256, 0, stream>>>(x, Wk, bk, kw, 0);
    qkv_proj_kernel<<<gemm_grid, 256, 0, stream>>>(x, Wv, bv, vw, 1);

    dim3 attn_grid(SEQ / 16, NH, BATCH);     // 128 x 16 x 2, one wave each
    attn_kernel<<<attn_grid, 32, 0, stream>>>(qw, kw, vw, aw);

    out_proj_kernel<<<gemm_grid, 256, 0, stream>>>(aw, Wo, bo, (float*)d_out);
}
